// DeformableEncoderBlock_51685636440861
// MI455X (gfx1250) — matrix operand order repaired and checked by reading, not compiled, not run on hardware
//
#include <hip/hip_runtime.h>

// ---------------------------------------------------------------------------
// Deformable DETR encoder block for gfx1250 (MI455X), bf16 WMMA GEMM pipeline.
// ---------------------------------------------------------------------------

typedef __bf16 bf16;
typedef __attribute__((ext_vector_type(16))) __bf16 v16bf;
typedef __attribute__((ext_vector_type(8)))  __bf16 v8bf;
typedef __attribute__((ext_vector_type(8)))  float  v8f;

#define BS 2
#define TOTAL 13294            // 100*100 + 50*50 + 25*25 + 13*13
#define ROWS (BS * TOTAL)      // 26588
#define EMBED 256
#define HEADS 8
#define HD 32
#define HID 1024

union ABFrag { v16bf v; v8bf h[2]; };

// ---------------------------------------------------------------------------
// f32 -> bf16 elementwise convert (weights + misc)
// ---------------------------------------------------------------------------
__global__ __launch_bounds__(256) void cvt_bf16_kernel(
    const float* __restrict__ in, bf16* __restrict__ out, int n)
{
    int i = blockIdx.x * 256 + threadIdx.x;
    if (i < n) out[i] = (bf16)in[i];
}

// ---------------------------------------------------------------------------
// LayerNorm over 256 columns; writes bf16 normalized row, optionally q = y+pos
// ---------------------------------------------------------------------------
__global__ __launch_bounds__(256) void ln_kernel(
    const float* __restrict__ x, const float* __restrict__ g,
    const float* __restrict__ be, const float* __restrict__ pos,
    bf16* __restrict__ xn, bf16* __restrict__ q)
{
    __shared__ float red[256];
    int t = threadIdx.x;
    size_t base = (size_t)blockIdx.x * EMBED;
    float v = x[base + t];
    red[t] = v; __syncthreads();
    #pragma unroll
    for (int s = 128; s > 0; s >>= 1) {
        if (t < s) red[t] += red[t + s];
        __syncthreads();
    }
    float mean = red[0] * (1.0f / EMBED);
    __syncthreads();
    float d = v - mean;
    red[t] = d * d; __syncthreads();
    #pragma unroll
    for (int s = 128; s > 0; s >>= 1) {
        if (t < s) red[t] += red[t + s];
        __syncthreads();
    }
    float var = red[0] * (1.0f / EMBED);
    float y = d * rsqrtf(var + 1e-5f) * g[t] + be[t];
    xn[base + t] = (bf16)y;
    if (q) q[base + t] = (bf16)(y + pos[base + t]);
}

// ---------------------------------------------------------------------------
// bf16 WMMA GEMM: C[M,N] = A[M,K] * B[K,N] + bias (+residual) (+relu)
// Block = 256 threads (8 waves), tile 128(M) x 64(N); each wave owns 32x32
// as a 2x2 grid of 16x16x32 v_wmma_f32_16x16x32_bf16 fragments.
// Output either f32 (Cf) or bf16 (Cb) — exactly one non-null.
// ---------------------------------------------------------------------------
__global__ __launch_bounds__(256) void gemm_bf16_kernel(
    const bf16* __restrict__ A, const bf16* __restrict__ B,
    const float* __restrict__ bias, const float* __restrict__ residual,
    float* __restrict__ Cf, bf16* __restrict__ Cb,
    int M, int N, int K, int relu)
{
    int lane  = threadIdx.x & 31;
    int wave  = threadIdx.x >> 5;     // 0..7
    int wm    = wave >> 1;            // 0..3 along M
    int wn    = wave & 1;             // 0..1 along N
    int row0  = blockIdx.y * 128 + wm * 32;
    int col0  = blockIdx.x * 64  + wn * 32;
    int lhalf = lane >> 4;            // 0/1
    int lmod  = lane & 15;

    v8f acc[2][2] = {};

    for (int k0 = 0; k0 < K; k0 += 32) {
        // ---- A fragments: lane holds row (row0+mi*16+lmod), two 16B chunks
        ABFrag a[2];
        #pragma unroll
        for (int mi = 0; mi < 2; ++mi) {
            int r = row0 + mi * 16 + lmod;
            if (r >= M) r = M - 1;                        // clamp; stores guarded
            const bf16* ap = A + (size_t)r * K + k0 + 8 * lhalf;
            a[mi].h[0] = *(const v8bf*)ap;                // K = base .. base+7
            a[mi].h[1] = *(const v8bf*)(ap + 16);         // K = base+16 .. +23
        }
        // ---- B fragments: lane holds column N=lmod, K = 8*lhalf+s (s<8), 16+8*lhalf+(s-8) (s>=8)
        ABFrag b[2];
        int kr = k0 + 8 * lhalf;
        #pragma unroll
        for (int ni = 0; ni < 2; ++ni) {
            const bf16* bp = B + (size_t)kr * N + col0 + ni * 16 + lmod;
            #pragma unroll
            for (int s = 0; s < 16; ++s)
                b[ni].v[s] = bp[(size_t)(s < 8 ? s : s + 8) * N];
        }
        // prefetch next K slab of A (global_prefetch_b8 path)
        if (k0 + 32 < K)
            __builtin_prefetch(A + (size_t)(row0 + lmod) * K + k0 + 32, 0, 1);

        #pragma unroll
        for (int mi = 0; mi < 2; ++mi)
            #pragma unroll
            for (int ni = 0; ni < 2; ++ni)
                acc[mi][ni] = __builtin_amdgcn_wmma_f32_16x16x32_bf16(
                    false, a[mi].v, false, b[ni].v,
                    (short)0, acc[mi][ni], false, false);
    }

    // ---- epilogue: VGPR r -> (M = r + 8*lhalf, N = lmod) within each frag
    #pragma unroll
    for (int mi = 0; mi < 2; ++mi) {
        #pragma unroll
        for (int ni = 0; ni < 2; ++ni) {
            #pragma unroll
            for (int r = 0; r < 8; ++r) {
                int m = row0 + mi * 16 + r + 8 * lhalf;
                int n = col0 + ni * 16 + lmod;
                if (m < M) {
                    float x = acc[mi][ni][r] + bias[n];
                    if (residual) x += residual[(size_t)m * N + n];
                    if (relu) x = fmaxf(x, 0.0f);
                    if (Cf) Cf[(size_t)m * N + n] = x;
                    else    Cb[(size_t)m * N + n] = (bf16)x;
                }
            }
        }
    }
}

// ---------------------------------------------------------------------------
// MSDeformAttn sampling: 1 wave per (b, t, head); lane = head-channel (HD=32).
// Softmax over LEVELS*POINTS=16 logits, 4-corner bilinear with per-corner
// validity masks (matches reference exactly). Output bf16 [R, 256].
// ---------------------------------------------------------------------------
__global__ __launch_bounds__(256) void msda_sample_kernel(
    const float* __restrict__ v,     // [R, 256]  (= [b*T+s, h*32+hd])
    const float* __restrict__ off,   // [R, 256]
    const float* __restrict__ aw,    // [R, 128]
    const float* __restrict__ ref,   // [R, 4, 2]
    bf16* __restrict__ attn_pre)     // [R, 256]
{
    const int HL[4] = {100, 50, 25, 13};
    const int WL[4] = {100, 50, 25, 13};
    const int ST[4] = {0, 10000, 12500, 13125};

    int lane = threadIdx.x & 31;
    int h    = threadIdx.x >> 5;          // 8 waves = 8 heads
    size_t bt = blockIdx.x;               // 0..R-1
    int b = (int)(bt / TOTAL);

    // softmax over 16 logits (replicated across lanes; hits L0/L2)
    const float* awp = aw + bt * 128 + h * 16;
    float w[16];
    float mx = -3.4e38f;
    #pragma unroll
    for (int j = 0; j < 16; ++j) { w[j] = awp[j]; mx = fmaxf(mx, w[j]); }
    float sum = 0.0f;
    #pragma unroll
    for (int j = 0; j < 16; ++j) { w[j] = __expf(w[j] - mx); sum += w[j]; }
    float inv = 1.0f / sum;

    const float* offp = off + bt * 256 + h * 32;   // [l*8 + p*2 + c]
    const float* refp = ref + bt * 8;              // [l*2 + c]

    float acc = 0.0f;
    #pragma unroll
    for (int l = 0; l < 4; ++l) {
        const int Hl = HL[l], Wl = WL[l];
        const float* vl = v + ((size_t)b * TOTAL + ST[l]) * 256 + h * 32 + lane;
        float rx = refp[l * 2 + 0], ry = refp[l * 2 + 1];
        #pragma unroll
        for (int p = 0; p < 4; ++p) {
            float x = rx * (float)Wl + offp[l * 8 + p * 2 + 0] - 0.5f;
            float y = ry * (float)Hl + offp[l * 8 + p * 2 + 1] - 0.5f;
            float x0f = floorf(x), y0f = floorf(y);
            float wx1 = x - x0f, wy1 = y - y0f;
            int x0 = (int)x0f, y0 = (int)y0f;

            auto samp = [&](int xi, int yi) -> float {
                bool valid = (xi >= 0) & (xi < Wl) & (yi >= 0) & (yi < Hl);
                int xc = min(max(xi, 0), Wl - 1);
                int yc = min(max(yi, 0), Hl - 1);
                float g = vl[(size_t)(yc * Wl + xc) * 256];
                return valid ? g : 0.0f;
            };
            float bil = (1.0f - wy1) * (1.0f - wx1) * samp(x0,     y0)
                      + (1.0f - wy1) * wx1          * samp(x0 + 1, y0)
                      + wy1          * (1.0f - wx1) * samp(x0,     y0 + 1)
                      + wy1          * wx1          * samp(x0 + 1, y0 + 1);
            acc += bil * (w[l * 4 + p] * inv);
        }
    }
    attn_pre[bt * 256 + h * 32 + lane] = (bf16)acc;
}

// ---------------------------------------------------------------------------
// Host-side orchestration
// ---------------------------------------------------------------------------
extern "C" void kernel_launch(void* const* d_in, const int* in_sizes, int n_in,
                              void* d_out, int out_size, void* d_ws, size_t ws_size,
                              hipStream_t stream)
{
    (void)in_sizes; (void)n_in; (void)out_size; (void)ws_size;

    const float* src     = (const float*)d_in[0];
    const float* pos     = (const float*)d_in[1];
    const float* ref     = (const float*)d_in[2];
    // d_in[3] spatial_shapes / d_in[4] level_start_index: compile-time constants
    const float* ln1_g   = (const float*)d_in[5];
    const float* ln1_b   = (const float*)d_in[6];
    const float* ln2_g   = (const float*)d_in[7];
    const float* ln2_b   = (const float*)d_in[8];
    const float* W_value = (const float*)d_in[9];
    const float* b_value = (const float*)d_in[10];
    const float* W_off   = (const float*)d_in[11];
    const float* b_off   = (const float*)d_in[12];
    const float* W_attn  = (const float*)d_in[13];
    const float* b_attn  = (const float*)d_in[14];
    const float* W_out   = (const float*)d_in[15];
    const float* b_out   = (const float*)d_in[16];
    const float* W_ffn1  = (const float*)d_in[17];
    const float* b_ffn1  = (const float*)d_in[18];
    const float* W_ffn2  = (const float*)d_in[19];
    const float* b_ffn2  = (const float*)d_in[20];
    float* out = (float*)d_out;

    const size_t R = ROWS;
    char* ws = (char*)d_ws;

    // workspace layout (bytes)
    size_t o = 0;
    bf16*  q_bf   = (bf16*) (ws + o); o += R * 256 * 2;   // q (bf16)
    bf16*  sn_bf  = (bf16*) (ws + o); o += R * 256 * 2;   // src_n (bf16)
    float* v_f    = (float*)(ws + o); o += R * 256 * 4;   // value proj (f32)
    float* off_f  = (float*)(ws + o); o += R * 256 * 4;   // offsets (f32)
    float* aw_f   = (float*)(ws + o); o += R * 128 * 4;   // attn logits (f32)
    bf16*  ap_bf  = (bf16*) (ws + o); o += R * 256 * 2;   // sampled attn (bf16)
    float* s1_f   = (float*)(ws + o); o += R * 256 * 4;   // src1 (f32)
    bf16*  Wv_bf  = (bf16*) (ws + o); o += 256 * 256 * 2;
    bf16*  Wo_bf  = (bf16*) (ws + o); o += 256 * 256 * 2;
    bf16*  Wa_bf  = (bf16*) (ws + o); o += 256 * 128 * 2;
    bf16*  Wp_bf  = (bf16*) (ws + o); o += 256 * 256 * 2;
    bf16*  W1_bf  = (bf16*) (ws + o); o += 256 * 1024 * 2;
    bf16*  W2_bf  = (bf16*) (ws + o); o += 1024 * 256 * 2;
    // reuse: FFN hidden h (R*1024 bf16 == R*256*8 bytes) overlays v_f+off_f
    bf16*  h_bf   = (bf16*)v_f;
    // reuse: src_n2 (R*256 bf16 == R*128*4 bytes) overlays aw_f
    bf16*  sn2_bf = (bf16*)aw_f;

    const int MB = (int)((R + 127) / 128);    // 208 M-tiles

    // 1) weights -> bf16
    auto cvt = [&](const float* s, bf16* d, int n) {
        cvt_bf16_kernel<<<(n + 255) / 256, 256, 0, stream>>>(s, d, n);
    };
    cvt(W_value, Wv_bf, 256 * 256);
    cvt(W_off,   Wo_bf, 256 * 256);
    cvt(W_attn,  Wa_bf, 256 * 128);
    cvt(W_out,   Wp_bf, 256 * 256);
    cvt(W_ffn1,  W1_bf, 256 * 1024);
    cvt(W_ffn2,  W2_bf, 1024 * 256);

    // 2) LN1: src_n (bf16) + q = src_n + pos (bf16)
    ln_kernel<<<(int)R, 256, 0, stream>>>(src, ln1_g, ln1_b, pos, sn_bf, q_bf);

    // 3) v = src_n @ W_value + b_value            [R,256]
    gemm_bf16_kernel<<<dim3(4, MB), 256, 0, stream>>>(
        sn_bf, Wv_bf, b_value, nullptr, v_f, nullptr, (int)R, 256, 256, 0);
    // 4) off = q @ W_off + b_off                  [R,256]
    gemm_bf16_kernel<<<dim3(4, MB), 256, 0, stream>>>(
        q_bf, Wo_bf, b_off, nullptr, off_f, nullptr, (int)R, 256, 256, 0);
    // 5) aw = q @ W_attn + b_attn                 [R,128]
    gemm_bf16_kernel<<<dim3(2, MB), 256, 0, stream>>>(
        q_bf, Wa_bf, b_attn, nullptr, aw_f, nullptr, (int)R, 128, 256, 0);

    // 6) deformable sampling -> attn_pre (bf16)   [R,256]
    msda_sample_kernel<<<(int)R, 256, 0, stream>>>(v_f, off_f, aw_f, ref, ap_bf);

    // 7) src1 = attn_pre @ W_out + b_out + src    [R,256]
    gemm_bf16_kernel<<<dim3(4, MB), 256, 0, stream>>>(
        ap_bf, Wp_bf, b_out, src, s1_f, nullptr, (int)R, 256, 256, 0);

    // 8) LN2 -> src_n2 (bf16)
    ln_kernel<<<(int)R, 256, 0, stream>>>(s1_f, ln2_g, ln2_b, nullptr, sn2_bf, nullptr);

    // 9) h = relu(src_n2 @ W_ffn1 + b_ffn1)       [R,1024] bf16
    gemm_bf16_kernel<<<dim3(16, MB), 256, 0, stream>>>(
        sn2_bf, W1_bf, b_ffn1, nullptr, nullptr, h_bf, (int)R, 1024, 256, 1);

    // 10) out = h @ W_ffn2 + b_ffn2 + src1        [R,256] f32 -> d_out
    gemm_bf16_kernel<<<dim3(4, MB), 256, 0, stream>>>(
        h_bf, W2_bf, b_ffn2, s1_f, out, nullptr, (int)R, 256, 1024, 0);
}
